// EdgeDegreeEmbedding_68839735820493
// MI455X (gfx1250) — matrix operand order repaired and checked by reading, not compiled, not run on hardware
//
#include <hip/hip_runtime.h>

typedef __attribute__((ext_vector_type(16))) _Float16 v16h;
typedef __attribute__((ext_vector_type(8)))  _Float16 v8h;
typedef __attribute__((ext_vector_type(8)))  float    v8f;

#define EIN   128   // edge_distance feature dim
#define HID   64    // hidden dim
#define OUTD  896   // M0 * SPHERE_C = 7 * 128
#define NUML  49    // (LMAX+1)^2
#define NUMM  29    // m-order count
#define SPHC  128
#define M0D   7

// LDS row strides (in halves) chosen for 16B alignment + conflict-free b128:
//  - 136 halves = 272B = 68 dwords : banks 4*l mod 64, distinct per 16 lanes
//  - 72  halves = 144B = 36 dwords : banks 36*l mod 64, 16 distinct 4-dword groups
#define W1S 136
#define W2S 72
#define W3S 72
#define HS  72

__device__ constexpr int JCOL[7] = {0, 2, 6, 11, 16, 21, 26};

// ---------------------------------------------------------------------------
// WMMA wrapper: D = A(16x32 f16) * B(32x16 f16) + C(16x16 f32)
// ---------------------------------------------------------------------------
__device__ inline v8f wmma16(v16h a, v16h b, v8f c) {
  return __builtin_amdgcn_wmma_f32_16x16x32_f16(
      /*neg_a=*/false, a, /*neg_b=*/false, b,
      /*c_mod=*/(short)0, c, /*reuse_a=*/false, /*reuse_b=*/false);
}

// 16 contiguous halves -> v16h (two b128 LDS/global loads)
__device__ inline v16h ld16h(const _Float16* p) {
  v8h lo = *(const v8h*)p;
  v8h hi = *(const v8h*)(p + 8);
  return __builtin_shufflevector(lo, hi, 0, 1, 2, 3, 4, 5, 6, 7,
                                 8, 9, 10, 11, 12, 13, 14, 15);
}

// halves at p[0..7] and p[16..23] -> v16h (A-fragment split-K pattern)
__device__ inline v16h ld16h_split(const _Float16* p) {
  v8h lo = *(const v8h*)p;
  v8h hi = *(const v8h*)(p + 16);
  return __builtin_shufflevector(lo, hi, 0, 1, 2, 3, 4, 5, 6, 7,
                                 8, 9, 10, 11, 12, 13, 14, 15);
}

// ---------------------------------------------------------------------------
// Zero the output buffer (harness poisons it with 0xAA).
// ---------------------------------------------------------------------------
__global__ void zero_out_kernel(float4* __restrict__ out, int n4) {
  int i = blockIdx.x * blockDim.x + threadIdx.x;
  if (i < n4) out[i] = make_float4(0.f, 0.f, 0.f, 0.f);
}

// ---------------------------------------------------------------------------
// LayerNorm + SiLU on 4 accumulated N-tiles (16 rows x 64 cols) in C/D layout,
// stored as f16 [16][HS] into LDS.
// C/D layout: VGPR v, lanes 0-15 -> (row v, col lane); lanes16-31 -> (row v+8).
// Row reduction: shfl_xor masks 1..8 (stay within the 16-lane half).
// ---------------------------------------------------------------------------
__device__ inline void ln_silu_to_lds(const v8f acc[4],
                                      const float* __restrict__ bias,
                                      const float* __restrict__ g,
                                      const float* __restrict__ be,
                                      _Float16* __restrict__ hw,
                                      int lh, int hi) {
  float bn[4], gn[4], ben[4];
#pragma unroll
  for (int nt = 0; nt < 4; ++nt) {
    int n = nt * 16 + lh;
    bn[nt] = bias[n]; gn[nt] = g[n]; ben[nt] = be[n];
  }
#pragma unroll
  for (int v = 0; v < 8; ++v) {
    float x[4];
    float s = 0.f, s2 = 0.f;
#pragma unroll
    for (int nt = 0; nt < 4; ++nt) {
      x[nt] = acc[nt][v] + bn[nt];
      s += x[nt]; s2 += x[nt] * x[nt];
    }
#pragma unroll
    for (int m = 1; m < 16; m <<= 1) {
      s  += __shfl_xor(s,  m, 32);
      s2 += __shfl_xor(s2, m, 32);
    }
    float mu  = s  * (1.f / HID);
    float var = s2 * (1.f / HID) - mu * mu;
    float inv = rsqrtf(var + 1e-5f);
    int row = v + (hi ? 8 : 0);
#pragma unroll
    for (int nt = 0; nt < 4; ++nt) {
      float y = (x[nt] - mu) * inv * gn[nt] + ben[nt];
      y = y / (1.f + __expf(-y));                 // SiLU
      hw[row * HS + nt * 16 + lh] = (_Float16)y;
    }
  }
}

// ---------------------------------------------------------------------------
// MLP kernel: one wave per 16-edge tile, 8 waves per block.
// Weights staged once per block into LDS as f16, TRANSPOSED (N-major, padded
// stride) so every WMMA B-fragment is two conflict-free ds_load_b128.
// Output: x_m0 as f16, layout [e][7][128].
// ---------------------------------------------------------------------------
__global__ __launch_bounds__(256)
void mlp_kernel(const float* __restrict__ xin,
                const float* __restrict__ w1, const float* __restrict__ b1,
                const float* __restrict__ g1, const float* __restrict__ be1,
                const float* __restrict__ w2, const float* __restrict__ b2,
                const float* __restrict__ g2, const float* __restrict__ be2,
                const float* __restrict__ w3, const float* __restrict__ b3,
                _Float16* __restrict__ xm0, int ntiles) {
  extern __shared__ _Float16 smem[];
  _Float16* w1t  = smem;                     // 64  rows * W1S (K=128)
  _Float16* w2t  = w1t + HID * W1S;          // 64  rows * W2S (K=64)
  _Float16* w3t  = w2t + HID * W2S;          // 896 rows * W3S (K=64)
  _Float16* hbuf = w3t + OUTD * W3S;         // 8 waves * 16 rows * HS

  // Cooperative f32 -> f16 transposed staging (global reads coalesced).
  for (int i = threadIdx.x; i < EIN * HID; i += 256) {
    int k = i >> 6, n = i & 63;              // w1 is (128,64)
    w1t[n * W1S + k] = (_Float16)w1[i];
  }
  for (int i = threadIdx.x; i < HID * HID; i += 256) {
    int k = i >> 6, n = i & 63;              // w2 is (64,64)
    w2t[n * W2S + k] = (_Float16)w2[i];
  }
  for (int i = threadIdx.x; i < HID * OUTD; i += 256) {
    int k = i / OUTD, n = i % OUTD;          // w3 is (64,896)
    w3t[n * W3S + k] = (_Float16)w3[i];
  }
  __syncthreads();

  int wave = threadIdx.x >> 5;
  int tile = blockIdx.x * 8 + wave;
  if (tile >= ntiles) return;
  int e0 = tile * 16;
  int l  = (int)(threadIdx.x & 31u);
  int lh = l & 15;            // lane within 16-lane half
  int hi = (l >> 4) & 1;      // which half
  _Float16* hw = hbuf + wave * 16 * HS;

  // -------- layer 1: (16x128) @ (128x64) --------
  v8f acc[4] = {};
#pragma unroll
  for (int kb = 0; kb < 4; ++kb) {
    // A fragment: lanes 0-15: K = kb*32 + {0..7,16..23}; lanes16-31: +8
    v16h a;
    {
      int kbase = kb * 32 + (hi ? 8 : 0);
      const float* p = xin + (size_t)(e0 + lh) * EIN + kbase;
#pragma unroll
      for (int i = 0; i < 8; ++i) a[i] = (_Float16)p[i];
#pragma unroll
      for (int i = 0; i < 8; ++i) a[8 + i] = (_Float16)p[16 + i];
    }
    int kbB = kb * 32 + (hi ? 8 : 0);
#pragma unroll
    for (int nt = 0; nt < 4; ++nt) {
      int col = nt * 16 + lh;
      v16h b = ld16h_split(w1t + (size_t)col * W1S + kbB);
      acc[nt] = wmma16(a, b, acc[nt]);
    }
  }
  ln_silu_to_lds(acc, b1, g1, be1, hw, lh, hi);

  // -------- layer 2: (16x64) @ (64x64) --------
  v8f acc2[4] = {};
#pragma unroll
  for (int kb = 0; kb < 2; ++kb) {
    v16h a = ld16h_split(hw + lh * HS + kb * 32 + (hi ? 8 : 0));
    int kbB = kb * 32 + (hi ? 8 : 0);
#pragma unroll
    for (int nt = 0; nt < 4; ++nt) {
      int col = nt * 16 + lh;
      v16h b = ld16h_split(w2t + (size_t)col * W2S + kbB);
      acc2[nt] = wmma16(a, b, acc2[nt]);
    }
  }
  ln_silu_to_lds(acc2, b2, g2, be2, hw, lh, hi);

  // -------- layer 3: (16x64) @ (64x896) --------
  v16h a3[2];
#pragma unroll
  for (int kb = 0; kb < 2; ++kb)
    a3[kb] = ld16h_split(hw + lh * HS + kb * 32 + (hi ? 8 : 0));

#pragma unroll 4
  for (int nt = 0; nt < 56; ++nt) {
    int col = nt * 16 + lh;
    v8f d = {};
#pragma unroll
    for (int kb = 0; kb < 2; ++kb) {
      v16h b = ld16h_split(w3t + (size_t)col * W3S + kb * 32 + (hi ? 8 : 0));
      d = wmma16(a3[kb], b, d);
    }
    float bias = b3[col];
#pragma unroll
    for (int v = 0; v < 8; ++v) {
      int row = v + (hi ? 8 : 0);
      xm0[(size_t)(e0 + row) * OUTD + col] = (_Float16)(d[v] + bias);
    }
  }
}

// ---------------------------------------------------------------------------
// Rotation + scatter: per edge, out[dst] += W[e][:,jcols] (49x7) @ x_m0 (7x128)
// One wave per edge; 4 M-tiles x 8 N-tiles WMMAs (K=7 padded into 32).
// Output (75 MB) is L2-resident (192MB L2), so atomics never touch HBM.
// ---------------------------------------------------------------------------
__global__ __launch_bounds__(256)
void rot_scatter_kernel(const float* __restrict__ wig,
                        const _Float16* __restrict__ xm0,
                        const int* __restrict__ edge_dst,
                        float* __restrict__ out, int E) {
  int e = blockIdx.x * 8 + (int)(threadIdx.x >> 5);
  if (e >= E) return;
  int l  = (int)(threadIdx.x & 31u);
  int lh = l & 15;
  int hi = (l >> 4) & 1;
  int dst = edge_dst[e];

  const float* we = wig + (size_t)e * NUML * NUMM;
  if (e + 8 < E) {
    // pull next edge's wigner block toward the caches
    __builtin_prefetch(wig + (size_t)(e + 8) * NUML * NUMM + l * 44, 0, 1);
  }

  // B fragments: x_m0[e] is (7 x 128) f16; pad K 7->32 (zeros).
  v16h bfr[8];
#pragma unroll
  for (int nt = 0; nt < 8; ++nt) {
    v16h b = {};
    if (!hi) {
      const _Float16* p = xm0 + (size_t)e * OUTD + nt * 16 + lh;
#pragma unroll
      for (int k = 0; k < 7; ++k) b[k] = p[k * SPHC];
    }
    bfr[nt] = b;
  }

  // A fragments: gathered wigner columns, scaled by 1/RESCALE, pad M 49->64.
  const float sA = (float)(1.0 / 23.395238876342773);
  v16h afr[4];
#pragma unroll
  for (int mt = 0; mt < 4; ++mt) {
    v16h a = {};
    int row = mt * 16 + lh;
    if (!hi && row < NUML) {
      const float* p = we + row * NUMM;
#pragma unroll
      for (int k = 0; k < 7; ++k) a[k] = (_Float16)(p[JCOL[k]] * sA);
    }
    afr[mt] = a;
  }

  float* ob = out + (size_t)dst * NUML * SPHC;
#pragma unroll
  for (int mt = 0; mt < 4; ++mt) {
    int rbase = mt * 16 + (hi ? 8 : 0);
#pragma unroll
    for (int nt = 0; nt < 8; ++nt) {
      v8f d = {};
      d = wmma16(afr[mt], bfr[nt], d);
      int col = nt * 16 + lh;
#pragma unroll
      for (int v = 0; v < 8; ++v) {
        int r = rbase + v;
        if (r < NUML) atomicAdd(ob + r * SPHC + col, d[v]);
      }
    }
  }
}

// ---------------------------------------------------------------------------
extern "C" void kernel_launch(void* const* d_in, const int* in_sizes, int n_in,
                              void* d_out, int out_size, void* d_ws, size_t ws_size,
                              hipStream_t stream) {
  const float* edge_distance = (const float*)d_in[0];
  const float* wigner        = (const float*)d_in[1];
  const float* w1  = (const float*)d_in[2];
  const float* b1  = (const float*)d_in[3];
  const float* g1  = (const float*)d_in[4];
  const float* be1 = (const float*)d_in[5];
  const float* w2  = (const float*)d_in[6];
  const float* b2  = (const float*)d_in[7];
  const float* g2  = (const float*)d_in[8];
  const float* be2 = (const float*)d_in[9];
  const float* w3  = (const float*)d_in[10];
  const float* b3  = (const float*)d_in[11];
  const int* edge_index = (const int*)d_in[12];

  int E = in_sizes[0] / EIN;
  _Float16* xm0 = (_Float16*)d_ws;   // E * 896 f16 (~107 MB)

  // 1) zero output
  int n4 = out_size / 4;
  hipLaunchKernelGGL(zero_out_kernel, dim3((n4 + 255) / 256), dim3(256), 0,
                     stream, (float4*)d_out, n4);

  // 2) MLP -> x_m0 (f16)
  int ntiles = (E + 15) / 16;
  int nblk   = (ntiles + 7) / 8;
  size_t smem = (size_t)(HID * W1S + HID * W2S + OUTD * W3S + 8 * 16 * HS)
                * sizeof(_Float16);   // ~170 KB of the 320 KB WGP LDS
  hipLaunchKernelGGL(mlp_kernel, dim3(nblk), dim3(256), smem, stream,
                     edge_distance, w1, b1, g1, be1, w2, b2, g2, be2, w3, b3,
                     xm0, ntiles);

  // 3) fused rotation + scatter-add (edge_index row 1 = destination)
  hipLaunchKernelGGL(rot_scatter_kernel, dim3((E + 7) / 8), dim3(256), 0,
                     stream, wigner, xm0, edge_index + E, (float*)d_out, E);
}
